// LearnedRouter_55860344652029
// MI455X (gfx1250) — matrix operand order repaired and checked by reading, not compiled, not run on hardware
//
#include <hip/hip_runtime.h>
#include <hip/hip_bf16.h>
#include <math.h>

typedef __bf16 bf16_t;
typedef __attribute__((ext_vector_type(8)))  __bf16 v8bf;
typedef __attribute__((ext_vector_type(16))) __bf16 v16bf;
typedef __attribute__((ext_vector_type(8)))  float  v8f;
typedef __attribute__((ext_vector_type(4)))  unsigned int v4u;
typedef __attribute__((ext_vector_type(8)))  int v8i;
typedef __attribute__((ext_vector_type(4)))  int v4i;

#define NEG_INF (-__builtin_huge_valf())

// Problem constants (from the reference)
constexpr int B_  = 4;
constexpr int S_  = 4096;
constexpr int M_  = 256;
constexpr int D_  = 1024;
constexpr int H_  = 16;
constexpr float SCALE_    = 0.125f;   // 1/sqrt(64)
constexpr float MIN_TEMP_ = 0.5f;

// d_out layout (flat, in return order, fp32)
constexpr size_t OFF_TOKEN = 0;                            // [B,S,D]
constexpr size_t OFF_BANK  = (size_t)B_ * S_ * D_;         // [B,S]
constexpr size_t OFF_W     = OFF_BANK + (size_t)B_ * S_;   // [B,H,S,M]
constexpr size_t OFF_TOPK  = OFF_W + (size_t)B_ * H_ * S_ * M_; // [B,H,S,8]

__device__ __forceinline__ v8f wmma_bf16(v16bf a, v16bf b, v8f c) {
    return __builtin_amdgcn_wmma_f32_16x16x32_bf16(false, a, false, b,
                                                   (short)0, c, false, false);
}

__device__ __forceinline__ bf16_t f2bf(float f) { return (bf16_t)f; }

// ---------------------------------------------------------------------------
// Kernel 0: zero-fill (float4 grid-stride) for the dense weights output
// ---------------------------------------------------------------------------
__global__ void zero_kernel(float4* __restrict__ p, size_t n4) {
    size_t i      = (size_t)blockIdx.x * blockDim.x + threadIdx.x;
    size_t stride = (size_t)gridDim.x * blockDim.x;
    float4 z; z.x = 0.f; z.y = 0.f; z.z = 0.f; z.w = 0.f;
    for (; i < n4; i += stride) p[i] = z;
}

// ---------------------------------------------------------------------------
// Kernel 1: GEMM + bias, fp32 in -> bf16 out.  out[r][n] = X[r][:] @ W[:][n] + b[n]
// Block tile 64(M) x 128(N), K-step 32, 256 threads = 8 waves, 4 WMMA tiles/wave.
// Grid: x = colBlock (8), y = rowBlock — column-blocks sharing an A-slice are
// adjacent in dispatch order so the A-slice stays L2-resident (A read ~once
// from HBM instead of 8x).
// ---------------------------------------------------------------------------
#define AS_STRIDE 40   // 64 rows x 32 k (bf16), stride mult of 8 for 16B ds loads
#define BS_STRIDE 40   // 128 cols x 32 k (bf16), column-major [n][k]

__global__ __launch_bounds__(256) void gemm_bias_bf16(
    const float* __restrict__ X, const float* __restrict__ W,
    const float* __restrict__ bias, bf16_t* __restrict__ out)
{
    __shared__ __align__(16) bf16_t As[64 * AS_STRIDE];
    __shared__ __align__(16) bf16_t Bs[128 * BS_STRIDE];

    const int tid     = threadIdx.x;
    const int colBase = blockIdx.x * 128;
    const int rowBase = blockIdx.y * 64;

    const int wave = tid >> 5;
    const int lane = tid & 31;
    const int l16  = lane & 15;
    const int hi   = lane >> 4;
    const int tr   = wave & 3;            // tile row 0..3 (16 rows each)
    const int tcg  = (wave >> 2) * 4;     // tile col group: cols tcg..tcg+3

    v8f acc[4] = {};

    // cooperative-load mappings
    const int a_row = tid >> 2;           // 0..63
    const int a_col = (tid & 3) * 8;      // 0,8,16,24
    const int b_n   = tid & 127;          // 0..127
    const int b_k   = (tid >> 7) * 16;    // 0 or 16

    for (int k0 = 0; k0 < D_; k0 += 32) {
        // A tile: 64 rows x 32 k (row-major), f32 -> bf16
        {
            const float* src = X + (size_t)(rowBase + a_row) * D_ + k0 + a_col;
            float4 f0 = *(const float4*)(src);
            float4 f1 = *(const float4*)(src + 4);
            __builtin_prefetch(src + 32, 0, 1);          // next k-step
            v8bf v;
            v[0] = f2bf(f0.x); v[1] = f2bf(f0.y); v[2] = f2bf(f0.z); v[3] = f2bf(f0.w);
            v[4] = f2bf(f1.x); v[5] = f2bf(f1.y); v[6] = f2bf(f1.z); v[7] = f2bf(f1.w);
            *(v8bf*)&As[a_row * AS_STRIDE + a_col] = v;
        }
        // B tile: 32 k x 128 n, stored column-major Bs[n][k]
        {
            const float* src = W + (size_t)(k0 + b_k) * D_ + colBase + b_n;
            __builtin_prefetch(src + (size_t)32 * D_, 0, 1);  // next k-step
            v8bf v0, v1;
            #pragma unroll
            for (int i = 0; i < 8; ++i)  v0[i] = f2bf(src[(size_t)i * D_]);
            #pragma unroll
            for (int i = 0; i < 8; ++i)  v1[i] = f2bf(src[(size_t)(i + 8) * D_]);
            *(v8bf*)&Bs[b_n * BS_STRIDE + b_k]     = v0;
            *(v8bf*)&Bs[b_n * BS_STRIDE + b_k + 8] = v1;
        }
        __syncthreads();

        // A fragment: lane holds row (tr*16 + l16); elements K = {8*hi.., 16+8*hi..}
        v16bf a;
        {
            const int r = tr * 16 + l16;
            v8bf s0 = *(const v8bf*)&As[r * AS_STRIDE + 8 * hi];
            v8bf s1 = *(const v8bf*)&As[r * AS_STRIDE + 16 + 8 * hi];
            #pragma unroll
            for (int e = 0; e < 8; ++e) { a[e] = s0[e]; a[e + 8] = s1[e]; }
        }
        #pragma unroll
        for (int j = 0; j < 4; ++j) {
            const int n = (tcg + j) * 16 + l16;
            v8bf t0 = *(const v8bf*)&Bs[n * BS_STRIDE + 8 * hi];
            v8bf t1 = *(const v8bf*)&Bs[n * BS_STRIDE + 16 + 8 * hi];
            v16bf bf;
            #pragma unroll
            for (int e = 0; e < 8; ++e) { bf[e] = t0[e]; bf[e + 8] = t1[e]; }
            acc[j] = wmma_bf16(a, bf, acc[j]);
        }
        __syncthreads();
    }

    // epilogue: C layout — VGPR j: M = j + 8*hi, N = l16
    #pragma unroll
    for (int j = 0; j < 4; ++j) {
        const int ncol = colBase + (tcg + j) * 16 + l16;
        const float bv = bias[ncol];
        const int rb   = rowBase + tr * 16 + 8 * hi;
        #pragma unroll
        for (int r = 0; r < 8; ++r)
            out[(size_t)(rb + r) * D_ + ncol] = f2bf(acc[j][r] + bv);
    }
}

// ---------------------------------------------------------------------------
// Kernel 2: per-(b,h) score tile [32 s x 256 m] via WMMA (K=64), then routing.
// The 256x64 bf16 K-tile is DMA'd into LDS by the Tensor Data Mover with the
// pad feature producing the 72-element row stride (pad_interval=32 DWORDs,
// pad_amount=4 DWORDs), per CDNA5 ISA.
// ---------------------------------------------------------------------------
__global__ __launch_bounds__(256) void route_kernel(
    const bf16_t* __restrict__ qbuf, const bf16_t* __restrict__ kbuf,
    const int* __restrict__ token_to_sets, const float* __restrict__ set_states,
    const float* __restrict__ temperature,
    float* __restrict__ out_token, float* __restrict__ out_weights,
    float* __restrict__ out_topk)
{
    // grid.x = B*H*(S/32) = 8192
    const int sblk = blockIdx.x & 127;         // S/32 = 128
    const int bh   = blockIdx.x >> 7;
    const int h    = bh & 15;
    const int b    = bh >> 4;
    const int s0   = sblk * 32;

    __shared__ __align__(16) bf16_t As[32 * 72];                 // 32 s-rows x 64 p
    __shared__ __align__(16) unsigned char BsMem[256 * 72 * 2];  // Bs, later scores
    bf16_t* Bs    = (bf16_t*)BsMem;                              // 256 m x 64 p (+pad)
    float* scores = (float*)BsMem;                               // 32 x 256 f32

    const int tid  = threadIdx.x;
    const int wave = tid >> 5;
    const int lane = tid & 31;
    const int l16  = lane & 15;
    const int hi   = lane >> 4;

    // ---- TDM: K tile (256 rows x 64 bf16, row stride 1024 elem) -> Bs ----
    if (wave == 0) {
        const unsigned long long gaddr =
            (unsigned long long)(const void*)(kbuf + (size_t)b * M_ * D_ + h * 64);
        const unsigned ldsa = (unsigned)(unsigned long long)(void*)BsMem;

        v4u g0;
        g0[0] = 1u;                         // count=1, user mode
        g0[1] = ldsa;                       // lds_addr (bytes)
        g0[2] = (unsigned)(gaddr & 0xFFFFFFFFu);
        g0[3] = (unsigned)((gaddr >> 32) & 0x1FFFFFFu) | (2u << 30); // type=2
        v8i g1;
        g1[0] = (int)((1u << 16)            // data_size = 2 bytes
                    | (1u << 20)            // pad_enable
                    | (4u << 22)            // pad_interval: 32 DWORDs (=128B row)
                    | (3u << 25));          // pad_amount: 4 DWORDs (=8 bf16)
        g1[1] = (int)(1024u << 16);         // tensor_dim0 = 1024 (low 16 in [31:16])
        g1[2] = (int)(256u << 16);          // dim0 hi16=0 | tensor_dim1 = 256
        g1[3] = (int)(64u << 16);           // dim1 hi16=0 | tile_dim0 = 64
        g1[4] = (int)256u;                  // tile_dim1 = 256 | tile_dim2 = 0
        g1[5] = (int)1024u;                 // tensor_dim0_stride = 1024 (lo32)
        g1[6] = 0;                          // stride0 hi16 | stride1 lo16
        g1[7] = 0;
        v4i g2 = {};                        // 2D: groups 2/3 zero
        v4i g3 = {};
        v8i g4 = {};                        // clang-23 6-arg form: extra group, zeros
        __builtin_amdgcn_tensor_load_to_lds(g0, g1, g2, g3, g4, 0);
    }

    // Stage q rows (already bf16) into LDS concurrently with the TDM copy
    {
        const int row = tid >> 3;              // 0..31
        const int seg = (tid & 7) * 8;         // 0..56
        v8bf v = *(const v8bf*)(qbuf + (size_t)(b * S_ + s0 + row) * D_ + h * 64 + seg);
        *(v8bf*)&As[row * 72 + seg] = v;
    }
    if (wave == 0) __builtin_amdgcn_s_wait_tensorcnt(0);
    __syncthreads();

    // WMMA — 2 s-tiles x 16 m-tiles, 8 waves x 4 tiles, K=64 (2 steps)
    const int tr  = wave & 1;
    const int tcg = (wave >> 1) * 4;
    v8f acc[4] = {};
    #pragma unroll
    for (int kk = 0; kk < 2; ++kk) {
        v16bf a;
        {
            const int r = tr * 16 + l16;
            v8bf s0v = *(const v8bf*)&As[r * 72 + kk * 32 + 8 * hi];
            v8bf s1v = *(const v8bf*)&As[r * 72 + kk * 32 + 16 + 8 * hi];
            #pragma unroll
            for (int e = 0; e < 8; ++e) { a[e] = s0v[e]; a[e + 8] = s1v[e]; }
        }
        #pragma unroll
        for (int j = 0; j < 4; ++j) {
            const int n = (tcg + j) * 16 + l16;
            v8bf t0 = *(const v8bf*)&Bs[n * 72 + kk * 32 + 8 * hi];
            v8bf t1 = *(const v8bf*)&Bs[n * 72 + kk * 32 + 16 + 8 * hi];
            v16bf bf;
            #pragma unroll
            for (int e = 0; e < 8; ++e) { bf[e] = t0[e]; bf[e + 8] = t1[e]; }
            acc[j] = wmma_bf16(a, bf, acc[j]);
        }
    }
    __syncthreads();   // all Bs reads done; safe to overwrite with scores

    #pragma unroll
    for (int j = 0; j < 4; ++j) {
        const int n = (tcg + j) * 16 + l16;
        #pragma unroll
        for (int r = 0; r < 8; ++r)
            scores[(tr * 16 + r + 8 * hi) * 256 + n] = acc[j][r] * SCALE_;
    }
    __syncthreads();

    const float temp = fmaxf(temperature[0], MIN_TEMP_);

    // Per-row routing; wave w handles rows w*4 .. w*4+3
    for (int rr = 0; rr < 4; ++rr) {
        const int r = wave * 4 + rr;
        const int s = s0 + r;

        const int mj  = token_to_sets[(size_t)s * 32 + lane];
        const bool valid = (mj >= 0);
        const int m   = valid ? (mj < M_ ? mj : M_ - 1) : 0;
        const unsigned key = valid ? (unsigned)m : 0xFFFFFFFFu;

        bool dup = false;
        for (int i = 0; i < 32; ++i) {
            unsigned ok = (unsigned)__shfl((int)key, i);
            if (i < lane && valid && ok == key) dup = true;
        }

        float cs  = (valid && !dup) ? scores[r * 256 + m] : NEG_INF;
        int   cm  = (valid && !dup) ? m : 0x7FFFFFFF;
        const int cslot = lane;

        float sel_sc[8]; int sel_m[8];
        #pragma unroll
        for (int it = 0; it < 8; ++it) {
            float bs = cs; int bm = cm; int bslot = cslot;
            #pragma unroll
            for (int off = 16; off > 0; off >>= 1) {
                float os = __shfl_xor(bs, off);
                int   om = __shfl_xor(bm, off);
                int   ot = __shfl_xor(bslot, off);
                if (os > bs || (os == bs && (om < bm || (om == bm && ot < bslot)))) {
                    bs = os; bm = om; bslot = ot;
                }
            }
            sel_sc[it] = bs; sel_m[it] = bm;
            if (bslot == cslot) cs = NEG_INF;   // remove winner
        }

        // softmax over the 8 selected (sel_sc[0] is the max)
        const float maxv = sel_sc[0];
        float w[8]; float sum = 0.f;
        #pragma unroll
        for (int k2 = 0; k2 < 8; ++k2) {
            w[k2] = (sel_sc[k2] > NEG_INF) ? expf((sel_sc[k2] - maxv) / temp) : 0.f;
            sum += w[k2];
        }
        const float inv = (sum > 0.f) ? 1.f / sum : 0.f;
        #pragma unroll
        for (int k2 = 0; k2 < 8; ++k2) w[k2] *= inv;

        // scatter topk_idx + weights (lane 0; all lanes hold identical sel/w)
        if (lane == 0) {
            const size_t rowi  = (size_t)(b * H_ + h) * S_ + s;
            const size_t base8 = rowi * 8;
            const size_t baseW = rowi * M_;
            #pragma unroll
            for (int k2 = 0; k2 < 8; ++k2) {
                out_topk[base8 + k2] = (float)sel_m[k2];
                if (sel_sc[k2] > NEG_INF) out_weights[baseW + sel_m[k2]] = w[k2];
            }
        }

        // weighted set combine: 64 dims, 2 per lane
        const int d0 = lane * 2;
        float a0 = 0.f, a1 = 0.f;
        #pragma unroll
        for (int k2 = 0; k2 < 8; ++k2) {
            if (sel_sc[k2] > NEG_INF) {
                const float* sp = set_states +
                    (size_t)(b * M_ + sel_m[k2]) * D_ + h * 64 + d0;
                a0 += w[k2] * sp[0];
                a1 += w[k2] * sp[1];
            }
        }
        float2 st; st.x = a0; st.y = a1;
        *(float2*)(out_token + (size_t)(b * S_ + s) * D_ + h * 64 + d0) = st;
    }
}

// ---------------------------------------------------------------------------
// Kernel 3: bank_indices = argmax_m mean_h weights  (deterministic recompute)
// one wave per (b,s); lane = candidate slot
// ---------------------------------------------------------------------------
__global__ __launch_bounds__(256) void bank_kernel(
    const float* __restrict__ wts, const int* __restrict__ token_to_sets,
    float* __restrict__ out_bank)
{
    const int wave = threadIdx.x >> 5;
    const int lane = threadIdx.x & 31;
    const int flat = blockIdx.x * 8 + wave;       // B*S = 16384
    const int b = flat >> 12;
    const int s = flat & (S_ - 1);

    const int mj  = token_to_sets[(size_t)s * 32 + lane];
    const bool valid = (mj >= 0);
    const int m   = valid ? (mj < M_ ? mj : M_ - 1) : 0;
    const unsigned key = valid ? (unsigned)m : 0xFFFFFFFFu;

    bool dup = false;
    for (int i = 0; i < 32; ++i) {
        unsigned ok = (unsigned)__shfl((int)key, i);
        if (i < lane && valid && ok == key) dup = true;
    }

    float sum = -1.f;
    if (valid && !dup) {
        sum = 0.f;
        #pragma unroll
        for (int hh = 0; hh < H_; ++hh)
            sum += wts[((size_t)(b * H_ + hh) * S_ + s) * M_ + m];
    }
    int bm = (valid && !dup) ? m : 0x7FFFFFFF;

    float bsum = sum;
    #pragma unroll
    for (int off = 16; off > 0; off >>= 1) {
        float os = __shfl_xor(bsum, off);
        int   om = __shfl_xor(bm, off);
        if (os > bsum || (os == bsum && om < bm)) { bsum = os; bm = om; }
    }
    if (lane == 0) out_bank[flat] = (float)(bm == 0x7FFFFFFF ? 0 : bm);
}

// ---------------------------------------------------------------------------
extern "C" void kernel_launch(void* const* d_in, const int* in_sizes, int n_in,
                              void* d_out, int out_size, void* d_ws, size_t ws_size,
                              hipStream_t stream) {
    (void)in_sizes; (void)n_in; (void)out_size; (void)ws_size;

    const float* token_states = (const float*)d_in[0];  // [B,S,D]
    const float* set_states   = (const float*)d_in[1];  // [B,M,D]
    const float* desc_router  = (const float*)d_in[2];  // [B,M,D]
    const int*   token_to_sets= (const int*)  d_in[3];  // [S,32]
    const float* Wq           = (const float*)d_in[4];  // [D,1024]
    const float* bq           = (const float*)d_in[5];
    const float* Wk           = (const float*)d_in[6];
    const float* bk           = (const float*)d_in[7];
    const float* temperature  = (const float*)d_in[8];

    float* out = (float*)d_out;

    // workspace: bf16 Q projection (33.5MB) + bf16 K projection (2MB)
    bf16_t* qbuf = (bf16_t*)d_ws;
    bf16_t* kbuf = (bf16_t*)((char*)d_ws + (size_t)B_ * S_ * D_ * sizeof(bf16_t));

    // 0) zero the dense weights output region
    {
        const size_t n4 = ((size_t)B_ * H_ * S_ * M_) / 4;
        zero_kernel<<<8192, 256, 0, stream>>>((float4*)(out + OFF_W), n4);
    }
    // 1) Q = token_states @ Wq + bq   (rows = B*S = 16384)
    gemm_bias_bf16<<<dim3(D_ / 128, (B_ * S_) / 64), 256, 0, stream>>>(
        token_states, Wq, bq, qbuf);
    // 2) K = desc_router @ Wk + bk    (rows = B*M = 1024)
    gemm_bias_bf16<<<dim3(D_ / 128, (B_ * M_) / 64), 256, 0, stream>>>(
        desc_router, Wk, bk, kbuf);
    // 3) scores (WMMA) + routing + combine
    route_kernel<<<B_ * H_ * (S_ / 32), 256, 0, stream>>>(
        qbuf, kbuf, token_to_sets, set_states, temperature,
        out + OFF_TOKEN, out + OFF_W, out + OFF_TOPK);
    // 4) bank indices
    bank_kernel<<<(B_ * S_) / 8, 256, 0, stream>>>(
        out + OFF_W, token_to_sets, out + OFF_BANK);
}
